// ReasonModule_37151467110480
// MI455X (gfx1250) — compile-verified, matrix operand order checked by reading, not run on hardware
//
#include <hip/hip_runtime.h>
#include <hip/hip_bf16.h>
#include <math.h>

// Problem constants (from reference setup_inputs)
#define CDIM 256
#define BDIM 512
#define LDIM 256
#define RDIM 8
#define KDIM 768          // 2C (q_star) + C (h) concatenated GEMM K
#define GDIM 1024         // 4C gate columns
#define PSTEPS 3

typedef __attribute__((ext_vector_type(16))) _Float16 v16h;
typedef __attribute__((ext_vector_type(8)))  _Float16 v8h;
typedef __attribute__((ext_vector_type(8)))  float    v8f;

// ---------------------------------------------------------------------------
// Kw: convert [w_ih | w_hh] -> f16 W[1024,768], bias = b_ih + b_hh
// ---------------------------------------------------------------------------
__global__ __launch_bounds__(256) void prep_weights(
    const float* __restrict__ w_ih, const float* __restrict__ w_hh,
    const float* __restrict__ b_ih, const float* __restrict__ b_hh,
    _Float16* __restrict__ W, float* __restrict__ bias) {
  int j = blockIdx.x;  // 0..1023 gate row
  for (int k = threadIdx.x; k < KDIM; k += 256) {
    float v = (k < 512) ? w_ih[(size_t)j * 512 + k]
                        : w_hh[(size_t)j * 256 + (k - 512)];
    W[(size_t)j * KDIM + k] = (_Float16)v;
  }
  if (threadIdx.x == 0) bias[j] = b_ih[j] + b_hh[j];
}

// ---------------------------------------------------------------------------
// K0b: f16 copy of input q_star into in0[:, 0:512]
// ---------------------------------------------------------------------------
__global__ __launch_bounds__(256) void prep_qstar(
    const float* __restrict__ qs, _Float16* __restrict__ in0) {
  int b = blockIdx.x;
  for (int j = threadIdx.x; j < 512; j += 256)
    in0[(size_t)b * KDIM + j] = (_Float16)qs[(size_t)b * 512 + j];
}

// ---------------------------------------------------------------------------
// K0: a_sit (h init) = bank_s[b,index,:] @ x[b,:,:]; cell state = 0;
//     also writes h as f16 into in0[:, 512:768]
// ---------------------------------------------------------------------------
__global__ __launch_bounds__(256) void init_h(
    const float* __restrict__ x, const float* __restrict__ bank_s,
    const int* __restrict__ index_p,
    float* __restrict__ h, float* __restrict__ cst,
    _Float16* __restrict__ in0) {
  __shared__ float s[LDIM];
  int b = blockIdx.x, tid = threadIdx.x;
  int idx = index_p[0];
  s[tid] = bank_s[(size_t)b * RDIM * LDIM + (size_t)idx * LDIM + tid];
  __syncthreads();
  const float* xb = x + (size_t)b * LDIM * CDIM;
  float acc = 0.f;
  for (int l = 0; l < LDIM; ++l) acc = fmaf(s[l], xb[(size_t)l * CDIM + tid], acc);
  h[(size_t)b * CDIM + tid]   = acc;
  cst[(size_t)b * CDIM + tid] = 0.f;
  in0[(size_t)b * KDIM + 512 + tid] = (_Float16)acc;
}

// ---------------------------------------------------------------------------
// K1: gates[512,1024] = in0f16[512,768] @ Wf16[1024,768]^T + bias
//     One wave per 16x16 tile, v_wmma_f32_16x16x32_f16, K=768 in 24 steps.
// ---------------------------------------------------------------------------
__global__ __launch_bounds__(256) void gates_gemm(
    const _Float16* __restrict__ in0, const _Float16* __restrict__ W,
    const float* __restrict__ bias, float* __restrict__ gates) {
  int wave = threadIdx.x >> 5;
  int lane = threadIdx.x & 31;
  int tile = blockIdx.x * 8 + wave;        // 2048 tiles = 32 (M) x 64 (N)
  int mi = tile & 31, ni = tile >> 5;
  int m0 = mi << 4, n0 = ni << 4;
  int lo = lane & 15, hi = lane >> 4;

  const _Float16* arow = in0 + (size_t)(m0 + lo) * KDIM;  // batch row
  const _Float16* brow = W   + (size_t)(n0 + lo) * KDIM;  // gate-col row (W^T access)

  v8f acc = {};
  for (int kb = 0; kb < KDIM; kb += 32) {
    v16h a, b;
    // ISA 16-bit A/B layout: VGPR0..3 -> K in [8*hi, 8*hi+8),
    //                        VGPR4..7 -> K in [16+8*hi, 16+8*hi+8)
    ((v8h*)&a)[0] = *(const v8h*)(arow + kb +      hi * 8);
    ((v8h*)&a)[1] = *(const v8h*)(arow + kb + 16 + hi * 8);
    ((v8h*)&b)[0] = *(const v8h*)(brow + kb +      hi * 8);
    ((v8h*)&b)[1] = *(const v8h*)(brow + kb + 16 + hi * 8);
    acc = __builtin_amdgcn_wmma_f32_16x16x32_f16(
        /*neg_a=*/false, a, /*neg_b=*/false, b,
        /*c_mod=*/(short)0, acc, /*reuse_a=*/false, /*reuse_b=*/false);
  }
  float bc = bias[n0 + lo];
#pragma unroll
  for (int v = 0; v < 8; ++v) {
    int m = m0 + v + (hi << 3);            // D: M = v + 8*hi, N = lane&15
    gates[(size_t)m * GDIM + n0 + lo] = acc[v] + bc;
  }
}

// ---------------------------------------------------------------------------
// K2: LSTM pointwise (torch gate order i,f,g,o); writes h (f32 + f16 into in0)
// ---------------------------------------------------------------------------
__global__ __launch_bounds__(256) void lstm_cell(
    const float* __restrict__ gates, float* __restrict__ cst,
    float* __restrict__ h, _Float16* __restrict__ in0) {
  int i = blockIdx.x * 256 + threadIdx.x;  // B*C = 131072
  int b = i >> 8, cc = i & 255;
  const float* g = gates + (size_t)b * GDIM;
  float ig = g[cc], fg = g[256 + cc], gg = g[512 + cc], og = g[768 + cc];
  float si = 1.f / (1.f + expf(-ig));
  float sf = 1.f / (1.f + expf(-fg));
  float so = 1.f / (1.f + expf(-og));
  float c = sf * cst[i] + si * tanhf(gg);
  float hn = so * tanhf(c);
  cst[i] = c;
  h[i] = hn;
  in0[(size_t)b * KDIM + 512 + cc] = (_Float16)hn;
}

// ---------------------------------------------------------------------------
// K3: fused segment attention softmax + pooling (one block per segment).
//     x streamed from L2 twice (dot pass + pooling pass).
//     Writes q_star f32 to qout and f16 [q|r|q] into in0 for next GEMM.
// ---------------------------------------------------------------------------
__global__ __launch_bounds__(256) void attn_pool(
    const float* __restrict__ x, const float* __restrict__ h,
    float* __restrict__ qout, _Float16* __restrict__ in0) {
  __shared__ float qsh[CDIM], ev[LDIM], red[256];
  int b = blockIdx.x, tid = threadIdx.x;
  int wave = tid >> 5, lane = tid & 31;
  qsh[tid] = h[(size_t)b * CDIM + tid];
  __syncthreads();
  const float* xb = x + (size_t)b * LDIM * CDIM;

  // e[l] = dot(x[b,l,:], q) -- wave-cooperative, coalesced 128B loads
  for (int l = wave; l < LDIM; l += 8) {
    const float* row = xb + (size_t)l * CDIM;
    float acc = 0.f;
#pragma unroll
    for (int i = 0; i < 8; ++i) {
      int c = lane + 32 * i;
      acc = fmaf(row[c], qsh[c], acc);
    }
    for (int off = 16; off; off >>= 1) acc += __shfl_xor(acc, off, 32);
    if (lane == 0) ev[l] = acc;
  }
  __syncthreads();

  // segment softmax: max, exp, sum, normalize
  red[tid] = ev[tid];
  __syncthreads();
  for (int s = 128; s; s >>= 1) {
    if (tid < s) red[tid] = fmaxf(red[tid], red[tid + s]);
    __syncthreads();
  }
  float mx = red[0];
  __syncthreads();
  float a = expf(ev[tid] - mx);
  ev[tid] = a;
  red[tid] = a;
  __syncthreads();
  for (int s = 128; s; s >>= 1) {
    if (tid < s) red[tid] += red[tid + s];
    __syncthreads();
  }
  float inv = 1.f / (red[0] + 1e-16f);
  __syncthreads();
  ev[tid] *= inv;
  __syncthreads();

  // r[c] = sum_l a[l] * x[b,l,c] -- coalesced across threads, LDS broadcast a[l]
  float acc = 0.f;
  for (int l = 0; l < LDIM; ++l)
    acc = fmaf(ev[l], xb[(size_t)l * CDIM + tid], acc);

  float q = qsh[tid];
  qout[(size_t)b * 512 + tid]       = q;
  qout[(size_t)b * 512 + 256 + tid] = acc;
  in0[(size_t)b * KDIM + tid]       = (_Float16)q;    // q_star[:, :256]
  in0[(size_t)b * KDIM + 256 + tid] = (_Float16)acc;  // q_star[:, 256:512]
  in0[(size_t)b * KDIM + 512 + tid] = (_Float16)q;    // h == q
}

// ---------------------------------------------------------------------------
extern "C" void kernel_launch(void* const* d_in, const int* in_sizes, int n_in,
                              void* d_out, int out_size, void* d_ws, size_t ws_size,
                              hipStream_t stream) {
  const float* x      = (const float*)d_in[0];
  // d_in[1] = batch: unused (segments are contiguous blocks of L rows)
  const float* q_star = (const float*)d_in[2];
  const float* bank_s = (const float*)d_in[3];
  const float* w_ih   = (const float*)d_in[4];
  const float* w_hh   = (const float*)d_in[5];
  const float* b_ih   = (const float*)d_in[6];
  const float* b_hh   = (const float*)d_in[7];
  const int*   index  = (const int*)d_in[8];
  float* out = (float*)d_out;

  // Workspace carve-up (256B aligned)
  char* ws = (char*)d_ws;
  size_t off = 0;
  auto carve = [&](size_t bytes) {
    char* p = ws + off;
    off = (off + bytes + 255) & ~(size_t)255;
    return p;
  };
  _Float16* Wf16  = (_Float16*)carve((size_t)GDIM * KDIM * 2);  // 1.5 MB
  _Float16* in0   = (_Float16*)carve((size_t)BDIM * KDIM * 2);  // 768 KB
  float*    gates = (float*)   carve((size_t)BDIM * GDIM * 4);  // 2 MB
  float*    h     = (float*)   carve((size_t)BDIM * CDIM * 4);  // 512 KB
  float*    cst   = (float*)   carve((size_t)BDIM * CDIM * 4);  // 512 KB
  float*    bias  = (float*)   carve((size_t)GDIM * 4);         // 4 KB
  (void)ws_size; (void)in_sizes; (void)n_in; (void)out_size;

  prep_weights<<<GDIM, 256, 0, stream>>>(w_ih, w_hh, b_ih, b_hh, Wf16, bias);
  prep_qstar  <<<BDIM, 256, 0, stream>>>(q_star, in0);
  init_h      <<<BDIM, 256, 0, stream>>>(x, bank_s, index, h, cst, in0);

  for (int t = 0; t < PSTEPS; ++t) {
    gates_gemm<<<256,  256, 0, stream>>>(in0, Wf16, bias, gates);
    lstm_cell <<<BDIM, 256, 0, stream>>>(gates, cst, h, in0);
    // writes q_star (f32) into d_out each step; final step's values remain
    attn_pool <<<BDIM, 256, 0, stream>>>(x, h, out, in0);
  }
}
